// MegNetBlock_v3_55851754717351
// MI455X (gfx1250) — compile-verified, matrix operand order checked by reading, not run on hardware
//
#include <hip/hip_runtime.h>

#define DIM 64
#define NNODES 50000
#define NEDGES 800000
#define NGRAPHS 1024
#define RS 256  // floats per activation row in LDS

typedef __attribute__((ext_vector_type(16))) __bf16 v16bf;
typedef __attribute__((ext_vector_type(8)))  __bf16 v8bf;
typedef __attribute__((ext_vector_type(8)))  float  v8f;
typedef __attribute__((ext_vector_type(4)))  float  v4f;

// may_alias views for in-place f32 <-> bf16 repacking in LDS
typedef v4f   av4f  __attribute__((may_alias));
typedef v8bf  av8bf __attribute__((may_alias));
typedef float af    __attribute__((may_alias));
typedef __bf16 abf  __attribute__((may_alias));

struct Layer { const float *W, *b, *g, *be; };

__device__ __forceinline__ void atomAddF(float* p, float v) {
  unsafeAtomicAdd(p, v);
}

// Stage W [K][N] row-major f32 (global) as W^T [N][K] bf16 (LDS). Coalesced reads.
__device__ __forceinline__ void stage_wT(__bf16* dst, const float* W, int K, int N, int tid) {
  for (int idx = tid; idx < K * N; idx += 256) {
    int k = idx / N, n = idx - k * N;
    ((abf*)dst)[n * K + k] = (__bf16)W[idx];
  }
}
__device__ __forceinline__ void stage_f(float* dst, const float* src, int n, int tid) {
  for (int i = tid; i < n; i += 256) dst[i] = src[i];
}

// LayerNorm (optionally of relu(prev)) from f32 at wbuf[row][src_off..src_off+K)
// into packed bf16 at start of each row. One lane per row; in-place-safe streaming.
template<int K>
__device__ __forceinline__ void wave_ln(float* wbuf, const float* g, const float* be,
                                        int src_off, bool relu, int lane) {
  if (lane < 16) {
    const float* src = wbuf + lane * RS + src_off;
    float s = 0.f, ss = 0.f;
#pragma unroll
    for (int i = 0; i < K; i += 4) {
      v4f v = *(const av4f*)(src + i);
#pragma unroll
      for (int j = 0; j < 4; ++j) {
        float xv = relu ? fmaxf(v[j], 0.f) : v[j];
        s += xv; ss += xv * xv;
      }
    }
    const float inv = 1.0f / (float)K;
    float mu = s * inv;
    float var = ss * inv - mu * mu;
    float rstd = rsqrtf(var + 1e-5f);
    abf* dst = (abf*)(wbuf + lane * RS);
#pragma unroll
    for (int i = 0; i < K; i += 4) {
      v4f v  = *(const av4f*)(src + i);
      v4f gg = *(const v4f*)(g + i);
      v4f bb = *(const v4f*)(be + i);
#pragma unroll
      for (int j = 0; j < 4; ++j) {
        float xv = relu ? fmaxf(v[j], 0.f) : v[j];
        dst[i + j] = (__bf16)((xv - mu) * rstd * gg[j] + bb[j]);
      }
    }
  }
}

// D[16,N] = A[16,K](bf16 act at row start) x W^T(LDS [N][K] bf16) + bias.
// D written f32 to wbuf[row][128 + n]. v_wmma_f32_16x16x32_bf16 chain.
template<int K, int N>
__device__ __forceinline__ void wave_gemm(float* wbuf, const __bf16* wT,
                                          const float* bias, int lane) {
  const int half = lane >> 4;
  const int l15  = lane & 15;
  const int koff = half * 8;
  const abf* arow = (const abf*)(wbuf + l15 * RS);
  v16bf a[K / 32];
#pragma unroll
  for (int kc = 0; kc < K / 32; ++kc) {
    v8bf lo = *(const av8bf*)(arow + kc * 32 + koff);
    v8bf hi = *(const av8bf*)(arow + kc * 32 + 16 + koff);
    a[kc] = __builtin_shufflevector(lo, hi, 0,1,2,3,4,5,6,7,8,9,10,11,12,13,14,15);
  }
#pragma unroll
  for (int nt = 0; nt < N / 16; ++nt) {
    const int col = nt * 16 + l15;
    const float bv = bias[col];
    v8f acc = { bv, bv, bv, bv, bv, bv, bv, bv };
    const abf* brow = (const abf*)(wT + col * K);
#pragma unroll
    for (int kc = 0; kc < K / 32; ++kc) {
      v8bf lo = *(const av8bf*)(brow + kc * 32 + koff);
      v8bf hi = *(const av8bf*)(brow + kc * 32 + 16 + koff);
      v16bf bm = __builtin_shufflevector(lo, hi, 0,1,2,3,4,5,6,7,8,9,10,11,12,13,14,15);
      acc = __builtin_amdgcn_wmma_f32_16x16x32_bf16(false, a[kc], false, bm,
                                                    (short)0, acc, false, false);
    }
#pragma unroll
    for (int i = 0; i < 8; ++i) {
      const int m = i + half * 8;
      ((af*)wbuf)[m * RS + 128 + nt * 16 + l15] = acc[i];
    }
  }
}

// ---------------- dense64: LN -> [64x128] -> ReLU -> LN -> [128x64] ----------
__global__ __launch_bounds__(256) void dense64_kernel(
    const float* __restrict__ in, float* __restrict__ out, int nrows,
    Layer L0, Layer L1) {
  extern __shared__ char smem[];
  __bf16* w0 = (__bf16*)smem;        // [128][64]
  __bf16* w1 = w0 + 128 * 64;        // [64][128]
  float* p   = (float*)(w1 + 64 * 128);
  float* act = p + 576;
  const int tid = threadIdx.x;
  stage_wT(w0, L0.W, 64, 128, tid);
  stage_wT(w1, L1.W, 128, 64, tid);
  stage_f(p +   0, L0.b, 128, tid);
  stage_f(p + 128, L0.g,  64, tid);
  stage_f(p + 192, L0.be, 64, tid);
  stage_f(p + 256, L1.b,  64, tid);
  stage_f(p + 320, L1.g, 128, tid);
  stage_f(p + 448, L1.be,128, tid);
  __syncthreads();
  const int wave = tid >> 5, lane = tid & 31;
  float* wbuf = act + wave * 16 * RS;
  const int r = lane & 15, half = lane >> 4;
  const int gr = blockIdx.x * 128 + wave * 16 + r;
  const bool valid = gr < nrows;
  const float* src = in + (long)(valid ? gr : 0) * 64 + half * 32;
#pragma unroll
  for (int i = 0; i < 32; i += 4)
    *(av4f*)(wbuf + r * RS + half * 32 + i) = *(const v4f*)(src + i);
  wave_ln<64>(wbuf, p + 128, p + 192, 0, false, lane);
  wave_gemm<64, 128>(wbuf, w0, p, lane);
  wave_ln<128>(wbuf, p + 320, p + 448, 128, true, lane);
  wave_gemm<128, 64>(wbuf, w1, p + 256, lane);
  if (valid) {
#pragma unroll
    for (int i = 0; i < 32; i += 4)
      *(v4f*)(out + (long)gr * 64 + half * 32 + i) =
          *(const av4f*)(wbuf + r * RS + 128 + half * 32 + i);
  }
}

// ---------------- edge: dense64(edge_attr) fused + gather + edge_msg ---------
__global__ __launch_bounds__(256) void edge_kernel(
    const int* __restrict__ ei, const float* __restrict__ eattr,
    const int* __restrict__ batch, const float* __restrict__ x_h,
    const float* __restrict__ u_h, float* __restrict__ edge_out,
    float* __restrict__ agg_sum, float* __restrict__ agg_cnt,
    float* __restrict__ e_gsum, float* __restrict__ e_gcnt,
    Layer d0, Layer d1, Layer m0, Layer m1, Layer m2) {
  extern __shared__ char smem[];
  __bf16* wd0 = (__bf16*)smem;        // [128][64]
  __bf16* wd1 = wd0 + 128 * 64;       // [64][128]
  __bf16* wm0 = wd1 + 64 * 128;       // [128][256]
  __bf16* wm1 = wm0 + 128 * 256;      // [128][128]
  __bf16* wm2 = wm1 + 128 * 128;      // [64][128]
  float* p   = (float*)(wm2 + 64 * 128);
  float* act = p + 1920;
  const int tid = threadIdx.x;
  stage_wT(wd0, d0.W, 64, 128, tid);
  stage_wT(wd1, d1.W, 128, 64, tid);
  stage_wT(wm0, m0.W, 256, 128, tid);
  stage_wT(wm1, m1.W, 128, 128, tid);
  stage_wT(wm2, m2.W, 128, 64, tid);
  stage_f(p +    0, d0.b, 128, tid);
  stage_f(p +  128, d0.g,  64, tid);
  stage_f(p +  192, d0.be, 64, tid);
  stage_f(p +  256, d1.b,  64, tid);
  stage_f(p +  320, d1.g, 128, tid);
  stage_f(p +  448, d1.be,128, tid);
  stage_f(p +  576, m0.b, 128, tid);
  stage_f(p +  704, m0.g, 256, tid);
  stage_f(p +  960, m0.be,256, tid);
  stage_f(p + 1216, m1.b, 128, tid);
  stage_f(p + 1344, m1.g, 128, tid);
  stage_f(p + 1472, m1.be,128, tid);
  stage_f(p + 1600, m2.b,  64, tid);
  stage_f(p + 1664, m2.g, 128, tid);
  stage_f(p + 1792, m2.be,128, tid);
  __syncthreads();
  const int wave = tid >> 5, lane = tid & 31;
  float* wbuf = act + wave * 16 * RS;
  const int r = lane & 15, half = lane >> 4;
  const int e = blockIdx.x * 128 + wave * 16 + r;
  const bool valid = e < NEDGES;
  const int es = valid ? e : 0;
  { // stage edge_attr
    const float* src = eattr + (long)es * 64 + half * 32;
#pragma unroll
    for (int i = 0; i < 32; i += 4)
      *(av4f*)(wbuf + r * RS + half * 32 + i) = *(const v4f*)(src + i);
  }
  // edge_dense: edge_h lands at floats [128..191] of each row
  wave_ln<64>(wbuf, p + 128, p + 192, 0, false, lane);
  wave_gemm<64, 128>(wbuf, wd0, p, lane);
  wave_ln<128>(wbuf, p + 320, p + 448, 128, true, lane);
  wave_gemm<128, 64>(wbuf, wd1, p + 256, lane);
  // gather e_in = [x_h[row] | x_h[col] | edge_h(in place) | u_h[batch[row]]]
  const int nrow = ei[es];
  const int ncol = ei[NEDGES + es];
  const int gid  = batch[nrow];
  {
    const float* s0 = x_h + (long)nrow * 64 + half * 32;
    const float* s1 = x_h + (long)ncol * 64 + half * 32;
    const float* s2 = u_h + (long)gid * 64 + half * 32;
#pragma unroll
    for (int i = 0; i < 32; i += 4) {
      *(av4f*)(wbuf + r * RS +       half * 32 + i) = *(const v4f*)(s0 + i);
      *(av4f*)(wbuf + r * RS +  64 + half * 32 + i) = *(const v4f*)(s1 + i);
      *(av4f*)(wbuf + r * RS + 192 + half * 32 + i) = *(const v4f*)(s2 + i);
    }
  }
  // edge_msg MLP
  wave_ln<256>(wbuf, p + 704, p + 960, 0, false, lane);
  wave_gemm<256, 128>(wbuf, wm0, p + 576, lane);
  wave_ln<128>(wbuf, p + 1344, p + 1472, 128, true, lane);
  wave_gemm<128, 128>(wbuf, wm1, p + 1216, lane);
  wave_ln<128>(wbuf, p + 1664, p + 1792, 128, true, lane);
  wave_gemm<128, 64>(wbuf, wm2, p + 1600, lane);
  if (valid) {
#pragma unroll
    for (int i = 0; i < 32; i += 4) {
      v4f d  = *(const av4f*)(wbuf + r * RS + 128 + half * 32 + i);
      v4f ea = *(const v4f*)(eattr + (long)e * 64 + half * 32 + i);
      v4f o;
#pragma unroll
      for (int j = 0; j < 4; ++j) o[j] = d[j] + ea[j];
      *(v4f*)(edge_out + (long)e * 64 + half * 32 + i) = o;
#pragma unroll
      for (int j = 0; j < 4; ++j) {
        atomAddF(agg_sum + (long)nrow * 64 + half * 32 + i + j, o[j]);
        atomAddF(e_gsum  + (long)gid  * 64 + half * 32 + i + j, o[j]);
      }
    }
    if (half == 0) {
      atomAddF(agg_cnt + nrow, 1.0f);
      atomAddF(e_gcnt + gid, 1.0f);
    }
  }
}

// ---------------- node: seg-mean + node_msg + residual ----------------------
__global__ __launch_bounds__(256) void node_kernel(
    const float* __restrict__ x, const int* __restrict__ batch,
    const float* __restrict__ x_h, const float* __restrict__ u_h,
    const float* __restrict__ agg_sum, const float* __restrict__ agg_cnt,
    float* __restrict__ x_out, float* __restrict__ n_gsum, float* __restrict__ n_gcnt,
    Layer m0, Layer m1, Layer m2) {
  extern __shared__ char smem[];
  __bf16* w0 = (__bf16*)smem;        // [128][192]
  __bf16* w1 = w0 + 128 * 192;       // [128][128]
  __bf16* w2 = w1 + 128 * 128;       // [64][128]
  float* p   = (float*)(w2 + 64 * 128);
  float* act = p + 1216;
  const int tid = threadIdx.x;
  stage_wT(w0, m0.W, 192, 128, tid);
  stage_wT(w1, m1.W, 128, 128, tid);
  stage_wT(w2, m2.W, 128, 64, tid);
  stage_f(p +   0, m0.b, 128, tid);
  stage_f(p + 128, m0.g, 192, tid);
  stage_f(p + 320, m0.be,192, tid);
  stage_f(p + 512, m1.b, 128, tid);
  stage_f(p + 640, m1.g, 128, tid);
  stage_f(p + 768, m1.be,128, tid);
  stage_f(p + 896, m2.b,  64, tid);
  stage_f(p + 960, m2.g, 128, tid);
  stage_f(p +1088, m2.be,128, tid);
  __syncthreads();
  const int wave = tid >> 5, lane = tid & 31;
  float* wbuf = act + wave * 16 * RS;
  const int r = lane & 15, half = lane >> 4;
  const int nd = blockIdx.x * 128 + wave * 16 + r;
  const bool valid = nd < NNODES;
  const int ns = valid ? nd : 0;
  const int gid = batch[ns];
  const float rc = 1.0f / fmaxf(agg_cnt[ns], 1.0f);
  {
    const float* s0 = agg_sum + (long)ns * 64 + half * 32;
    const float* s1 = x_h + (long)ns * 64 + half * 32;
    const float* s2 = u_h + (long)gid * 64 + half * 32;
#pragma unroll
    for (int i = 0; i < 32; i += 4) {
      v4f a0 = *(const v4f*)(s0 + i);
#pragma unroll
      for (int j = 0; j < 4; ++j) a0[j] *= rc;
      *(av4f*)(wbuf + r * RS +       half * 32 + i) = a0;
      *(av4f*)(wbuf + r * RS +  64 + half * 32 + i) = *(const v4f*)(s1 + i);
      *(av4f*)(wbuf + r * RS + 128 + half * 32 + i) = *(const v4f*)(s2 + i);
    }
  }
  wave_ln<192>(wbuf, p + 128, p + 320, 0, false, lane);
  wave_gemm<192, 128>(wbuf, w0, p, lane);
  wave_ln<128>(wbuf, p + 640, p + 768, 128, true, lane);
  wave_gemm<128, 128>(wbuf, w1, p + 512, lane);
  wave_ln<128>(wbuf, p + 960, p + 1088, 128, true, lane);
  wave_gemm<128, 64>(wbuf, w2, p + 896, lane);
  if (valid) {
#pragma unroll
    for (int i = 0; i < 32; i += 4) {
      v4f d  = *(const av4f*)(wbuf + r * RS + 128 + half * 32 + i);
      v4f xv = *(const v4f*)(x + (long)nd * 64 + half * 32 + i);
      v4f o;
#pragma unroll
      for (int j = 0; j < 4; ++j) o[j] = d[j] + xv[j];
      *(v4f*)(x_out + (long)nd * 64 + half * 32 + i) = o;
#pragma unroll
      for (int j = 0; j < 4; ++j)
        atomAddF(n_gsum + (long)gid * 64 + half * 32 + i + j, o[j]);
    }
    if (half == 0) atomAddF(n_gcnt + gid, 1.0f);
  }
}

// ---------------- global: seg-means + global_msg + residual -----------------
__global__ __launch_bounds__(256) void global_kernel(
    const float* __restrict__ u, const float* __restrict__ u_h,
    const float* __restrict__ n_gsum, const float* __restrict__ n_gcnt,
    const float* __restrict__ e_gsum, const float* __restrict__ e_gcnt,
    float* __restrict__ u_out, Layer m0, Layer m1, Layer m2) {
  extern __shared__ char smem[];
  __bf16* w0 = (__bf16*)smem;
  __bf16* w1 = w0 + 128 * 192;
  __bf16* w2 = w1 + 128 * 128;
  float* p   = (float*)(w2 + 64 * 128);
  float* act = p + 1216;
  const int tid = threadIdx.x;
  stage_wT(w0, m0.W, 192, 128, tid);
  stage_wT(w1, m1.W, 128, 128, tid);
  stage_wT(w2, m2.W, 128, 64, tid);
  stage_f(p +   0, m0.b, 128, tid);
  stage_f(p + 128, m0.g, 192, tid);
  stage_f(p + 320, m0.be,192, tid);
  stage_f(p + 512, m1.b, 128, tid);
  stage_f(p + 640, m1.g, 128, tid);
  stage_f(p + 768, m1.be,128, tid);
  stage_f(p + 896, m2.b,  64, tid);
  stage_f(p + 960, m2.g, 128, tid);
  stage_f(p +1088, m2.be,128, tid);
  __syncthreads();
  const int wave = tid >> 5, lane = tid & 31;
  float* wbuf = act + wave * 16 * RS;
  const int r = lane & 15, half = lane >> 4;
  const int g = blockIdx.x * 128 + wave * 16 + r;
  const bool valid = g < NGRAPHS;
  const int gs = valid ? g : 0;
  const float rn = 1.0f / fmaxf(n_gcnt[gs], 1.0f);
  const float re = 1.0f / fmaxf(e_gcnt[gs], 1.0f);
  {
    const float* s0 = u_h   + (long)gs * 64 + half * 32;
    const float* s1 = n_gsum + (long)gs * 64 + half * 32;
    const float* s2 = e_gsum + (long)gs * 64 + half * 32;
#pragma unroll
    for (int i = 0; i < 32; i += 4) {
      v4f a1 = *(const v4f*)(s1 + i);
      v4f a2 = *(const v4f*)(s2 + i);
#pragma unroll
      for (int j = 0; j < 4; ++j) { a1[j] *= rn; a2[j] *= re; }
      *(av4f*)(wbuf + r * RS +       half * 32 + i) = *(const v4f*)(s0 + i);
      *(av4f*)(wbuf + r * RS +  64 + half * 32 + i) = a1;
      *(av4f*)(wbuf + r * RS + 128 + half * 32 + i) = a2;
    }
  }
  wave_ln<192>(wbuf, p + 128, p + 320, 0, false, lane);
  wave_gemm<192, 128>(wbuf, w0, p, lane);
  wave_ln<128>(wbuf, p + 640, p + 768, 128, true, lane);
  wave_gemm<128, 128>(wbuf, w1, p + 512, lane);
  wave_ln<128>(wbuf, p + 960, p + 1088, 128, true, lane);
  wave_gemm<128, 64>(wbuf, w2, p + 896, lane);
  if (valid) {
#pragma unroll
    for (int i = 0; i < 32; i += 4) {
      v4f d  = *(const av4f*)(wbuf + r * RS + 128 + half * 32 + i);
      v4f uv = *(const v4f*)(u + (long)g * 64 + half * 32 + i);
      v4f o;
#pragma unroll
      for (int j = 0; j < 4; ++j) o[j] = d[j] + uv[j];
      *(v4f*)(u_out + (long)g * 64 + half * 32 + i) = o;
    }
  }
}

__global__ void zero_kernel(float* __restrict__ ptr, long n) {
  long i = (long)blockIdx.x * blockDim.x + threadIdx.x;
  const long stride = (long)gridDim.x * blockDim.x;
  for (; i < n; i += stride) ptr[i] = 0.0f;
}

static constexpr int SMEM_DENSE = (128*64 + 64*128) * 2 + 576 * 4 + 8 * 16 * RS * 4;
static constexpr int SMEM_EDGE  = (128*64 + 64*128 + 128*256 + 128*128 + 64*128) * 2
                                  + 1920 * 4 + 8 * 16 * RS * 4;
static constexpr int SMEM_MSG   = (128*192 + 128*128 + 64*128) * 2 + 1216 * 4
                                  + 8 * 16 * RS * 4;

extern "C" void kernel_launch(void* const* d_in, const int* in_sizes, int n_in,
                              void* d_out, int out_size, void* d_ws, size_t ws_size,
                              hipStream_t stream) {
  (void)in_sizes; (void)n_in; (void)out_size; (void)ws_size;
  const float* x     = (const float*)d_in[0];
  const int*   ei    = (const int*)d_in[1];
  const float* eattr = (const float*)d_in[2];
  const float* u     = (const float*)d_in[3];
  const int*   batch = (const int*)d_in[4];
  auto L = [&](int i) {
    Layer l{ (const float*)d_in[i], (const float*)d_in[i+1],
             (const float*)d_in[i+2], (const float*)d_in[i+3] };
    return l;
  };
  Layer ed0 = L(5),  ed1 = L(9);
  Layer nd0 = L(13), nd1 = L(17);
  Layer gd0 = L(21), gd1 = L(25);
  Layer em0 = L(29), em1 = L(33), em2 = L(37);
  Layer nm0 = L(41), nm1 = L(45), nm2 = L(49);
  Layer gm0 = L(53), gm1 = L(57), gm2 = L(61);

  float* ws      = (float*)d_ws;
  float* x_h     = ws;                       // 3,200,000
  float* u_h     = x_h + (long)NNODES * 64;  // 65,536
  float* agg_sum = u_h + (long)NGRAPHS * 64; // 3,200,000
  float* agg_cnt = agg_sum + (long)NNODES * 64;
  float* e_gsum  = agg_cnt + NNODES;
  float* e_gcnt  = e_gsum + (long)NGRAPHS * 64;
  float* n_gsum  = e_gcnt + NGRAPHS;
  float* n_gcnt  = n_gsum + (long)NGRAPHS * 64;

  float* x_out    = (float*)d_out;
  float* edge_out = x_out + (long)NNODES * 64;
  float* u_out    = edge_out + (long)NEDGES * 64;

  (void)hipFuncSetAttribute((const void*)dense64_kernel,
      hipFuncAttributeMaxDynamicSharedMemorySize, SMEM_DENSE);
  (void)hipFuncSetAttribute((const void*)edge_kernel,
      hipFuncAttributeMaxDynamicSharedMemorySize, SMEM_EDGE);
  (void)hipFuncSetAttribute((const void*)node_kernel,
      hipFuncAttributeMaxDynamicSharedMemorySize, SMEM_MSG);
  (void)hipFuncSetAttribute((const void*)global_kernel,
      hipFuncAttributeMaxDynamicSharedMemorySize, SMEM_MSG);

  const long nz = (long)NNODES * 64 + NNODES + (long)NGRAPHS * 64 + NGRAPHS
                + (long)NGRAPHS * 64 + NGRAPHS;
  zero_kernel<<<2048, 256, 0, stream>>>(agg_sum, nz);

  dense64_kernel<<<(NNODES + 127) / 128, 256, SMEM_DENSE, stream>>>(
      x, x_h, NNODES, nd0, nd1);
  dense64_kernel<<<(NGRAPHS + 127) / 128, 256, SMEM_DENSE, stream>>>(
      u, u_h, NGRAPHS, gd0, gd1);
  edge_kernel<<<(NEDGES + 127) / 128, 256, SMEM_EDGE, stream>>>(
      ei, eattr, batch, x_h, u_h, edge_out,
      agg_sum, agg_cnt, e_gsum, e_gcnt,
      ed0, ed1, em0, em1, em2);
  node_kernel<<<(NNODES + 127) / 128, 256, SMEM_MSG, stream>>>(
      x, batch, x_h, u_h, agg_sum, agg_cnt,
      x_out, n_gsum, n_gcnt, nm0, nm1, nm2);
  global_kernel<<<(NGRAPHS + 127) / 128, 256, SMEM_MSG, stream>>>(
      u, u_h, n_gsum, n_gcnt, e_gsum, e_gcnt,
      u_out, gm0, gm1, gm2);
}
